// GlowTTS_8907762172587
// MI455X (gfx1250) — matrix operand order repaired and checked by reading, not compiled, not run on hardware
//
#include <hip/hip_runtime.h>
#include <hip/hip_bf16.h>
#include <math.h>

typedef __attribute__((ext_vector_type(16))) __bf16 v16bf;
typedef __attribute__((ext_vector_type(8)))  float  v8f;

#define NEGV (-1e9f)

static __device__ __forceinline__ unsigned short f32_to_bf16(float f) {
    unsigned int u = __float_as_uint(f);
    unsigned int r = (u + 0x7FFFu + ((u >> 16) & 1u)) >> 16;
    return (unsigned short)r;
}

// ---------------------------------------------------------------------------
// Pack text-frame features directly in WMMA *B-operand* fragment order:
//   B 32x16 fragment: element (k, n) -> lane = ((k>>4)<<4) | n, h = k & 15
// AfragG layout: [B][tTile][kk][lane(32)][h(16)] ushort (32B per lane).
// Also tvec[b][t] = logp1 + logp4 scalar terms and dur_log passthrough.
// ---------------------------------------------------------------------------
__global__ void pack_t_kernel(const float* __restrict__ o_mean,
                              const float* __restrict__ o_log_scale,
                              const float* __restrict__ o_dur_log,
                              unsigned short* __restrict__ AfragG,
                              float* __restrict__ tvec,
                              float* __restrict__ out_durlog,
                              int B, int C, int Tx, int ksteps, int nTTp) {
    int i = blockIdx.x * blockDim.x + threadIdx.x;
    if (i >= B * Tx) return;
    int b = i / Tx, t = i % Tx;
    const float HALF_LOG_2PI = 0.9189385332046727f;
    float acc = 0.f;
    int C2 = 2 * C;
    for (int g16 = 0; g16 < ksteps * 2; ++g16) {
        int kk = g16 >> 1, g2 = g16 & 1;
        union { unsigned short u[16]; uint4 q[2]; } vals;
#pragma unroll
        for (int e = 0; e < 16; ++e) {
            int f = g16 * 16 + e;            // feature index = K coordinate
            unsigned short v = 0;
            if (f < C2) {
                int c = (f < C) ? f : f - C;
                float ls = o_log_scale[((size_t)b * C + c) * Tx + t];
                float mu = o_mean[((size_t)b * C + c) * Tx + t];
                float sc = expf(-2.f * ls);
                if (f < C) {
                    v = f32_to_bf16(sc);
                    acc += -HALF_LOG_2PI - ls - 0.5f * mu * mu * sc;
                } else {
                    v = f32_to_bf16(mu * sc);
                }
            }
            vals.u[e] = v;
        }
        int lane = (g2 << 4) | (t & 15);
        size_t base = ((((size_t)b * nTTp + (t >> 4)) * ksteps + kk) * 32 + lane) * 16;
        uint4* dst = (uint4*)(AfragG + base);
        dst[0] = vals.q[0];
        dst[1] = vals.q[1];
    }
    tvec[i] = acc;
    out_durlog[i] = o_dur_log[i];   // [B,1,Tx] -> [B,Tx,1] is a flat copy
}

// ---------------------------------------------------------------------------
// Pack mel-frame features in WMMA *A-operand* fragment order:
//   A 16x32 fragment: element (m, k) -> lane = ((k>>4)<<4) | m,
//                                       h = k & 15
// ZfragG layout: [B][sTile][kk][lane(32)][h(16)] ushort.
// ---------------------------------------------------------------------------
__global__ void pack_s_kernel(const float* __restrict__ z,
                              const int* __restrict__ y_lengths,
                              unsigned short* __restrict__ ZfragG,
                              int B, int C, int Ty, int ksteps, int nSTp) {
    int i = blockIdx.x * blockDim.x + threadIdx.x;
    if (i >= B * Ty) return;
    int b = i / Ty, s = i % Ty;
    int ylen2 = (y_lengths[b] >> 1) << 1;
    float ym = (s < ylen2) ? 1.f : 0.f;
    int C2 = 2 * C;
    for (int o = 0; o < ksteps * 4; ++o) {
        int kk = o >> 2, k0 = (o & 3) * 8;
        int g  = k0 >> 4;
        int h0 = k0 & 8;
        union { unsigned short u[8]; uint4 q; } vals;
#pragma unroll
        for (int e = 0; e < 8; ++e) {
            int f = kk * 32 + k0 + e;
            unsigned short v = 0;
            if (f < C2) {
                int c = (f < C) ? f : f - C;
                float zv = z[((size_t)b * C + c) * Ty + s] * ym;
                v = f32_to_bf16((f < C) ? (-0.5f * zv * zv) : zv);
            }
            vals.u[e] = v;
        }
        int lane = (g << 4) | (s & 15);
        size_t base = ((((size_t)b * nSTp + (s >> 4)) * ksteps + kk) * 32 + lane) * 16 + h0;
        *(uint4*)(ZfragG + base) = vals.q;
    }
}

// ---------------------------------------------------------------------------
// Shared epilogue: + per-t scalar terms, attn mask, transposed store.
// ---------------------------------------------------------------------------
static __device__ __forceinline__ void gemm_epilogue(
        v8f acc[2][4], const float* __restrict__ tvec,
        const int* __restrict__ x_lengths, const int* __restrict__ y_lengths,
        float* __restrict__ logp, int b, int s0, int t0, int lane,
        int Tx, int Ty) {
    int xlen  = x_lengths[b];
    int ylen2 = (y_lengths[b] >> 1) << 1;
    int n = lane & 15, g = lane >> 4;
#pragma unroll
    for (int i = 0; i < 2; ++i) {
#pragma unroll
        for (int j = 0; j < 4; ++j) {
            int tcol = t0 + 16 * j + n;
            float tv = (tcol < Tx) ? tvec[(size_t)b * Tx + tcol] : 0.f;
#pragma unroll
            for (int r = 0; r < 8; ++r) {     // C/D: m = r + 8*g, n = lane&15
                int s = s0 + 16 * i + r + 8 * g;
                if (s < Ty && tcol < Tx) {
                    float val = acc[i][j][r] + tv;
                    if (tcol >= xlen || s >= ylen2) val = NEGV;
                    logp[((size_t)b * Ty + s) * Tx + tcol] = val;
                }
            }
        }
    }
}

// ---------------------------------------------------------------------------
// logp[b][s][t] = sum_k Zfeat[s][k] * Afeat[k][t] + tvec[t], masked to NEG.
// Fragments pre-swizzled in global memory: one 32B load per lane per fragment
// (2x global_load_b128 with constant offsets), reuse across tiles via L2.
// Wave tile 32(s) x 64(t) = 2x4 WMMA accumulators.
// KSTEPS compile-time => fully static 2-stage software pipeline: no runtime
// buffer index, no movrels, no scratch spills.
// ---------------------------------------------------------------------------
template<int KSTEPS>
__global__ __launch_bounds__(128)
void gemm_logp_tpl(const unsigned short* __restrict__ AfragG,
                   const unsigned short* __restrict__ ZfragG,
                   const float* __restrict__ tvec,
                   const int* __restrict__ x_lengths,
                   const int* __restrict__ y_lengths,
                   float* __restrict__ logp,     // [B][Ty][Tx]
                   int B, int Tx, int Ty, int nSTp, int nTTp) {
    int b    = blockIdx.z;
    int lane = threadIdx.x & 31;
    int wave = threadIdx.x >> 5;
    int t0   = blockIdx.x * 64;
    int s0   = blockIdx.y * 128 + wave * 32;
    if (s0 >= Ty) return;

    size_t zb = ((size_t)b * nSTp + (s0 >> 4)) * KSTEPS;
    size_t ab = ((size_t)b * nTTp + (t0 >> 4)) * KSTEPS;
    const char* zbase = (const char*)ZfragG + (zb * 32 + lane) * 32;  // 32B/lane
    const char* abase = (const char*)AfragG + (ab * 32 + lane) * 32;

#define ZFRAG(i, kk) (*(const v16bf*)(zbase + ((i) * KSTEPS + (kk)) * 1024))
#define AFRAG(j, kk) (*(const v16bf*)(abase + ((j) * KSTEPS + (kk)) * 1024))

    v8f   acc[2][4] = {};
    v16bf zaA[2], baA[4], zaB[2], baB[4];
#pragma unroll
    for (int i = 0; i < 2; ++i) zaA[i] = ZFRAG(i, 0);
#pragma unroll
    for (int j = 0; j < 4; ++j) baA[j] = AFRAG(j, 0);

#pragma unroll
    for (int k2 = 0; k2 < (KSTEPS + 1) / 2; ++k2) {
        const int kB = 2 * k2 + 1;
        if (kB < KSTEPS) {            // prefetch odd step into buffer B
#pragma unroll
            for (int i = 0; i < 2; ++i) zaB[i] = ZFRAG(i, kB);
#pragma unroll
            for (int j = 0; j < 4; ++j) baB[j] = AFRAG(j, kB);
        }
#pragma unroll
        for (int i = 0; i < 2; ++i)
#pragma unroll
            for (int j = 0; j < 4; ++j)
                acc[i][j] = __builtin_amdgcn_wmma_f32_16x16x32_bf16(
                    false, zaA[i], false, baA[j], (short)0, acc[i][j],
                    false, false);
        if (kB + 1 < KSTEPS) {        // prefetch next even step into buffer A
#pragma unroll
            for (int i = 0; i < 2; ++i) zaA[i] = ZFRAG(i, kB + 1);
#pragma unroll
            for (int j = 0; j < 4; ++j) baA[j] = AFRAG(j, kB + 1);
        }
        if (kB < KSTEPS) {
#pragma unroll
            for (int i = 0; i < 2; ++i)
#pragma unroll
                for (int j = 0; j < 4; ++j)
                    acc[i][j] = __builtin_amdgcn_wmma_f32_16x16x32_bf16(
                        false, zaB[i], false, baB[j], (short)0, acc[i][j],
                        false, false);
        }
    }
#undef ZFRAG
#undef AFRAG

    gemm_epilogue(acc, tvec, x_lengths, y_lengths, logp, b, s0, t0, lane, Tx, Ty);
}

// Generic fallback for shapes where ksteps != 5: simple (non-pipelined) loop,
// fresh fragment variables per iteration (no runtime buffer indexing).
__global__ __launch_bounds__(128)
void gemm_logp_generic(const unsigned short* __restrict__ AfragG,
                       const unsigned short* __restrict__ ZfragG,
                       const float* __restrict__ tvec,
                       const int* __restrict__ x_lengths,
                       const int* __restrict__ y_lengths,
                       float* __restrict__ logp,
                       int B, int ksteps, int Tx, int Ty, int nSTp, int nTTp) {
    int b    = blockIdx.z;
    int lane = threadIdx.x & 31;
    int wave = threadIdx.x >> 5;
    int t0   = blockIdx.x * 64;
    int s0   = blockIdx.y * 128 + wave * 32;
    if (s0 >= Ty) return;

    size_t zb = ((size_t)b * nSTp + (s0 >> 4)) * ksteps;
    size_t ab = ((size_t)b * nTTp + (t0 >> 4)) * ksteps;
    const char* zbase = (const char*)ZfragG + (zb * 32 + lane) * 32;
    const char* abase = (const char*)AfragG + (ab * 32 + lane) * 32;

    v8f acc[2][4] = {};
    for (int kk = 0; kk < ksteps; ++kk) {
        v16bf za[2], ba[4];
#pragma unroll
        for (int i = 0; i < 2; ++i)
            za[i] = *(const v16bf*)(zbase + ((size_t)(i * ksteps + kk)) * 1024);
#pragma unroll
        for (int j = 0; j < 4; ++j)
            ba[j] = *(const v16bf*)(abase + ((size_t)(j * ksteps + kk)) * 1024);
#pragma unroll
        for (int i = 0; i < 2; ++i)
#pragma unroll
            for (int j = 0; j < 4; ++j)
                acc[i][j] = __builtin_amdgcn_wmma_f32_16x16x32_bf16(
                    false, za[i], false, ba[j], (short)0, acc[i][j],
                    false, false);
    }
    gemm_epilogue(acc, tvec, x_lengths, y_lengths, logp, b, s0, t0, lane, Tx, Ty);
}

// ---------------------------------------------------------------------------
// Viterbi forward (in-place logp -> DP values), then serial backtrack writing
// one-hot rows of attn^T, duration counts, and the gather index buffer.
// One workgroup per batch, blockDim == Tx (<=512).
// ---------------------------------------------------------------------------
__global__ void dp_path_kernel(float* __restrict__ values,      // [B][Ty][Tx]
                               const int* __restrict__ x_lengths,
                               const int* __restrict__ y_lengths,
                               float* __restrict__ attnT,       // [B][Ty][Tx]
                               float* __restrict__ out_attndur, // [B][Tx]
                               int* __restrict__ idx_buf,       // [B][Ty]
                               int Tx, int Ty) {
    __shared__ float vbuf[2][512];
    __shared__ int   counts[512];
    __shared__ int   sh_idx;

    int b = blockIdx.x;
    int t = threadIdx.x;
    int xlen  = x_lengths[b];
    int ylen2 = (y_lengths[b] >> 1) << 1;
    float* base = values + (size_t)b * Ty * Tx;

    vbuf[0][t] = (t == 0) ? 0.f : NEGV;
    counts[t]  = 0;
    if (t == 0) sh_idx = xlen - 1;
    __syncthreads();

    int cur = 0;
    for (int j = 0; j < Ty; ++j) {
        float col = base[(size_t)j * Tx + t];
        if (j + 1 < Ty) __builtin_prefetch(&base[(size_t)(j + 1) * Tx + t], 0, 1);
        float left = (t == 0) ? NEGV : vbuf[cur][t - 1];
        float self = vbuf[cur][t];
        float vn = col + fmaxf(self, left);
        base[(size_t)j * Tx + t] = vn;      // DP table in-place
        vbuf[cur ^ 1][t] = vn;
        __syncthreads();
        cur ^= 1;
    }

    float* attb = attnT + (size_t)b * Ty * Tx;
    for (int j = Ty - 1; j >= 0; --j) {
        int idx = sh_idx;
        int active = (j < ylen2);
        attb[(size_t)j * Tx + t] = (active && t == idx) ? 1.f : 0.f;
        __syncthreads();             // everyone read sh_idx before update
        if (t == 0) {
            idx_buf[b * Ty + j] = active ? idx : -1;
            if (active) {
                counts[idx] += 1;
                if (j > 0 && idx != 0) {
                    float vh = base[(size_t)(j - 1) * Tx + idx];
                    float vl = base[(size_t)(j - 1) * Tx + idx - 1];
                    if (idx == j || vh < vl) sh_idx = idx - 1;
                }
            }
        }
        __syncthreads();
    }
    out_attndur[(size_t)b * Tx + t] = (t < xlen) ? log1pf((float)counts[t]) : 0.f;
}

// ---------------------------------------------------------------------------
// One-hot alignment => output einsums become gathers.
// ---------------------------------------------------------------------------
__global__ void gather_kernel(const float* __restrict__ z,
                              const float* __restrict__ o_mean,
                              const float* __restrict__ o_log_scale,
                              const int* __restrict__ y_lengths,
                              const int* __restrict__ idx_buf,
                              float* __restrict__ out_z,
                              float* __restrict__ out_mean,
                              float* __restrict__ out_ls,
                              int B, int C, int Tx, int Ty) {
    long long i = blockIdx.x * (long long)blockDim.x + threadIdx.x;
    long long total = (long long)B * Ty * C;
    if (i >= total) return;
    int c = (int)(i % C);
    long long r = i / C;
    int s = (int)(r % Ty);
    int b = (int)(r / Ty);
    int ylen2 = (y_lengths[b] >> 1) << 1;
    float ym = (s < ylen2) ? 1.f : 0.f;
    out_z[i] = z[((size_t)b * C + c) * Ty + s] * ym;
    int idx = idx_buf[b * Ty + s];
    if (idx >= 0) {
        out_mean[i] = o_mean[((size_t)b * C + c) * Tx + idx];
        out_ls[i]   = o_log_scale[((size_t)b * C + c) * Tx + idx];
    } else {
        out_mean[i] = 0.f;
        out_ls[i]   = 0.f;
    }
}

// ---------------------------------------------------------------------------
extern "C" void kernel_launch(void* const* d_in, const int* in_sizes, int n_in,
                              void* d_out, int out_size, void* d_ws, size_t ws_size,
                              hipStream_t stream) {
    const float* o_mean      = (const float*)d_in[0];
    const float* o_log_scale = (const float*)d_in[1];
    const float* o_dur_log   = (const float*)d_in[2];
    const float* z           = (const float*)d_in[3];
    const int*   x_lengths   = (const int*)d_in[4];
    const int*   y_lengths   = (const int*)d_in[5];

    int B  = in_sizes[4];                 // x_lengths: [B]
    int Tx = in_sizes[2] / B;             // o_dur_log: [B,1,Tx]
    int C  = in_sizes[0] / (B * Tx);      // o_mean:    [B,C,Tx]
    int Ty = in_sizes[3] / (B * C);       // z:         [B,C,Ty]

    int ksteps = (2 * C + 31) / 32;       // K tiles of 32 (2C=160 -> 5)
    int nblkX  = (Tx + 63) / 64;
    int nblkY  = (Ty + 127) / 128;
    int nTTp   = nblkX * 4;               // padded 16-wide t tiles
    int nSTp   = nblkY * 8;               // padded 16-wide s tiles

    float* out = (float*)d_out;
    size_t szZ = (size_t)B * Ty * C;
    float* out_z       = out;                             // [B,Ty,C]
    float* out_mean    = out + szZ;                       // [B,Ty,C]
    float* out_ls      = out + 2 * szZ;                   // [B,Ty,C]
    float* out_attn    = out + 3 * szZ;                   // [B,Ty,Tx]
    float* out_durlog  = out_attn + (size_t)B * Ty * Tx;  // [B,Tx,1]
    float* out_attndur = out_durlog + (size_t)B * Tx;     // [B,Tx,1]

    char* ws = (char*)d_ws;
    size_t off = 0;
    auto carve = [&](size_t bytes) -> void* {
        void* p = ws + off;
        off = (off + bytes + 255) & ~(size_t)255;
        return p;
    };
    float*          logp    = (float*)carve((size_t)B * Ty * Tx * 4);
    unsigned short* AfragG  = (unsigned short*)carve((size_t)B * nTTp * ksteps * 1024);
    unsigned short* ZfragG  = (unsigned short*)carve((size_t)B * nSTp * ksteps * 1024);
    float*          tvec    = (float*)carve((size_t)B * Tx * 4);
    int*            idx_buf = (int*)carve((size_t)B * Ty * 4);
    (void)ws_size; (void)n_in; (void)out_size;

    // 1) pack bf16 operands in WMMA fragment order (+ scalar terms, dur_log)
    pack_t_kernel<<<(B * Tx + 255) / 256, 256, 0, stream>>>(
        o_mean, o_log_scale, o_dur_log, AfragG, tvec, out_durlog,
        B, C, Tx, ksteps, nTTp);
    pack_s_kernel<<<(B * Ty + 255) / 256, 256, 0, stream>>>(
        z, y_lengths, ZfragG, B, C, Ty, ksteps, nSTp);

    // 2) fused logp GEMM: bf16 WMMA, f32 acc, masked, transposed store
    dim3 ggrid(nblkX, nblkY, B);
    if (ksteps == 5) {
        gemm_logp_tpl<5><<<ggrid, 128, 0, stream>>>(
            AfragG, ZfragG, tvec, x_lengths, y_lengths, logp,
            B, Tx, Ty, nSTp, nTTp);
    } else {
        gemm_logp_generic<<<ggrid, 128, 0, stream>>>(
            AfragG, ZfragG, tvec, x_lengths, y_lengths, logp,
            B, ksteps, Tx, Ty, nSTp, nTTp);
    }

    // 3) Viterbi DP + backtrack (one block per batch)
    int dpthreads = (Tx < 512) ? Tx : 512;
    dp_path_kernel<<<B, dpthreads, 0, stream>>>(
        logp, x_lengths, y_lengths, out_attn, out_attndur, idx_buf, Tx, Ty);

    // 4) output gathers
    long long total = (long long)B * Ty * C;
    gather_kernel<<<(unsigned)((total + 255) / 256), 256, 0, stream>>>(
        z, o_mean, o_log_scale, y_lengths, idx_buf,
        out_z, out_mean, out_ls, B, C, Tx, Ty);
}
